// DonutSwinSelfAttention_84679575208289
// MI455X (gfx1250) — compile-verified, matrix operand order checked by reading, not run on hardware
//
#include <hip/hip_runtime.h>
#include <hip/hip_bf16.h>

// ---- problem constants -----------------------------------------------------
#define WIN      7
#define NTOK     49          // tokens per window
#define NPAD     64          // padded tokens (4 tiles of 16)
#define DIM      512
#define HEADS    16
#define HD       32          // head dim == one f16 WMMA K-step
#define NWIN     4096
#define WPB      2           // windows per block
#define SCALE    0.17677669529663687f   // HD^-0.5

typedef __attribute__((ext_vector_type(16))) _Float16 v16h;
typedef __attribute__((ext_vector_type(8)))  float    v8f;

// Offset (in halves) of element (m, k) inside a 16x32 f16 A-operand tile
// (per CDNA5 ISA 7.12.2). The B-operand (32x16) layout is the mirror with
// m := n, so a tile packed this way for M reads as M^T when fed as B.
__device__ __forceinline__ int a_off(int m, int k) {
    int lane = (m & 15) + (((k >> 3) & 1) << 4);
    int h    = (k & 7) + ((k >> 4) << 3);
    return lane * 16 + h;
}

// ---- prep 1: pack Wq/Wk/Wv (f32 [512,512] row-major in->out) into f16
//      A-operand tiles of W^T: Wp[mat][ntile(32)][ktile(16)][512 halves].
//      Tile element (n, k) = W[kt*32+k][ntile*16+n].
__global__ void pack_weights(const float* __restrict__ Wq,
                             const float* __restrict__ Wk,
                             const float* __restrict__ Wv,
                             _Float16* __restrict__ Wp) {
    int i = blockIdx.x * 256 + threadIdx.x;          // 3*32*16*512 = 786432
    if (i >= 3 * 32 * 16 * 512) return;
    int within = i & 511;
    int tile   = i >> 9;
    int lane   = within >> 4;
    int hh     = within & 15;
    int kt     = tile & 15;
    int ntile  = (tile >> 4) & 31;
    int mat    = tile >> 9;
    int ncol   = ntile * 16 + (lane & 15);
    int kl     = (hh & 7) + (((lane >> 4) & 1) << 3) + ((hh >> 3) << 4);
    int k      = kt * 32 + kl;
    const float* W = (mat == 0) ? Wq : ((mat == 1) ? Wk : Wv);
    Wp[i] = (_Float16)W[k * DIM + ncol];
}

// ---- prep 2: expand rel-pos bias to padded [HEADS][64][64] f32 -------------
__global__ void expand_bias(const float* __restrict__ table,
                            const int* __restrict__ idx,
                            float* __restrict__ biasE) {
    int i = blockIdx.x * 256 + threadIdx.x;          // 16*64*64 = 65536
    if (i >= HEADS * NPAD * NPAD) return;
    int h  = i >> 12;
    int q  = (i >> 6) & 63;
    int kc = i & 63;
    float v = -1e30f;                                // padding -> exp() == 0
    if (q < NTOK && kc < NTOK)
        v = table[idx[q * NTOK + kc] * HEADS + h];
    biasE[i] = v;
}

// ---- fused window attention ------------------------------------------------
// grid = NWIN/WPB blocks, 256 threads (8 waves). Dynamic LDS:
//   [0,128K)   packed X A-tiles: [WPB][4 mt][16 kt][512 halves]
//   [128K,..)  per-wave staging (7168 halves): Q[4*512] K[4*512] V[4*512] P[2*512]
__global__ void __launch_bounds__(256)
swin_attn(const float* __restrict__ X,
          const float* __restrict__ bq,
          const float* __restrict__ bk,
          const float* __restrict__ bv,
          const _Float16* __restrict__ Wp,
          const float* __restrict__ biasE,
          float* __restrict__ out) {
    extern __shared__ char smem[];
    _Float16* Xp    = (_Float16*)smem;
    _Float16* stage = (_Float16*)(smem + WPB * 4 * 16 * 512 * 2);

    const int tid  = threadIdx.x;
    const int wave = tid >> 5;
    const int lane = tid & 31;
    const int ncol = lane & 15;       // D-layout column / token-local in D'
    const int hi   = (lane >> 4) & 1;

    // Phase 1: load X (f32), convert, pack into A-operand layout in LDS.
    const float* Xg = X + (size_t)blockIdx.x * WPB * NTOK * DIM;
    for (int e = tid; e < WPB * NPAD * DIM; e += 256) {
        int w = e >> 15, rem = e & 32767;
        int m = rem >> 9, k = rem & 511;
        float v = (m < NTOK) ? Xg[(w * NTOK + m) * DIM + k] : 0.0f;
        int off = ((w * 4 + (m >> 4)) * 16 + (k >> 5)) * 512 + a_off(m & 15, k & 31);
        Xp[off] = (_Float16)v;
    }
    __syncthreads();

    _Float16* st = stage + wave * 7168;   // private per-wave staging
    const v8f zero = {};

    for (int p = wave; p < WPB * HEADS; p += 8) {
        const int w = p >> 4;
        const int h = p & 15;

        // ---- QKV projection, transposed: D' = W_tile^T (A) x X^T (B) ------
        // One weight A-tile (1 KB from L2) is reused across the 4 token
        // tiles; X streams from LDS. D'(n, m) = (X@W)(m, n).
        for (int mat = 0; mat < 3; ++mat) {
            const float* bias = (mat == 0) ? bq : ((mat == 1) ? bk : bv);
            for (int nt = 0; nt < 2; ++nt) {
                v8f acc[4] = {zero, zero, zero, zero};
                for (int kt = 0; kt < 16; ++kt) {
                    v16h Aw = *(const v16h*)(Wp +
                        (((mat * 32 + (h * 2 + nt)) * 16 + kt) * 512) + lane * 16);
                    for (int mt = 0; mt < 4; ++mt) {
                        v16h Bx = *(const v16h*)(Xp +
                            ((w * 4 + mt) * 16 + kt) * 512 + lane * 16);
                        acc[mt] = __builtin_amdgcn_wmma_f32_16x16x32_f16(
                            false, Aw, false, Bx, (short)0, acc[mt], false, false);
                    }
                }
                // finalize + stage. D' tile: lane holds token mloc = lane&15,
                // head-dim d = nt*16 + r + 8*hi across the 8 accum regs.
                float bb[8];
                for (int r = 0; r < 8; ++r)
                    bb[r] = bias[h * HD + nt * 16 + r + hi * 8];
                for (int mt = 0; mt < 4; ++mt) {
                    for (int r = 0; r < 8; ++r) {
                        int d = nt * 16 + r + hi * 8;  // 0..31 within head
                        float v = acc[mt][r] + bb[r];
                        if (mat == 0) {
                            st[mt * 512 + a_off(ncol, d)] = (_Float16)(v * SCALE);
                        } else if (mat == 1) {
                            st[2048 + mt * 512 + a_off(ncol, d)] = (_Float16)v;
                        } else {
                            int tok = mt * 16 + ncol;
                            st[4096 + ((tok >> 5) * 2 + nt) * 512 +
                               a_off(d & 15, tok & 31)] = (_Float16)v;
                        }
                    }
                }
            }
        }

        // ---- attention: scores -> softmax -> context ----------------------
        const float* bh = biasE + h * NPAD * NPAD;
        for (int mt = 0; mt < 4; ++mt) {
            v16h Qt = *(const v16h*)(st + mt * 512 + lane * 16);
            v8f sc[4];
            for (int nt = 0; nt < 4; ++nt) {
                // K staged in A-layout == B-operand layout of K^T
                v16h Kt = *(const v16h*)(st + 2048 + nt * 512 + lane * 16);
                sc[nt] = __builtin_amdgcn_wmma_f32_16x16x32_f16(
                    false, Qt, false, Kt, (short)0, zero, false, false);
            }
            // softmax per row (row = 16 lanes of one VGPR in the D layout)
            for (int r = 0; r < 8; ++r) {
                int q = mt * 16 + r + hi * 8;
                float vals[4];
                float mx = -3.0e38f;
                for (int nt = 0; nt < 4; ++nt) {
                    vals[nt] = sc[nt][r] + bh[q * NPAD + nt * 16 + ncol];
                    mx = fmaxf(mx, vals[nt]);
                }
                mx = fmaxf(mx, __shfl_xor(mx, 1));
                mx = fmaxf(mx, __shfl_xor(mx, 2));
                mx = fmaxf(mx, __shfl_xor(mx, 4));
                mx = fmaxf(mx, __shfl_xor(mx, 8));
                float s = 0.0f;
                for (int nt = 0; nt < 4; ++nt) {
                    float e = __expf(vals[nt] - mx);
                    vals[nt] = e;
                    s += e;
                }
                s += __shfl_xor(s, 1);
                s += __shfl_xor(s, 2);
                s += __shfl_xor(s, 4);
                s += __shfl_xor(s, 8);
                float inv = 1.0f / s;
                for (int nt = 0; nt < 4; ++nt) {
                    float pv = vals[nt] * inv;
                    int kc = nt * 16 + ncol;
                    st[6144 + (kc >> 5) * 512 + a_off(r + hi * 8, kc & 31)] =
                        (_Float16)pv;
                }
            }
            // context: P[16,64] @ V[64,32]
            v8f ct[2] = {zero, zero};
            for (int ktt = 0; ktt < 2; ++ktt) {
                v16h Pt = *(const v16h*)(st + 6144 + ktt * 512 + lane * 16);
                for (int ntd = 0; ntd < 2; ++ntd) {
                    v16h Vt = *(const v16h*)(st + 4096 + (ktt * 2 + ntd) * 512 + lane * 16);
                    ct[ntd] = __builtin_amdgcn_wmma_f32_16x16x32_f16(
                        false, Pt, false, Vt, (short)0, ct[ntd], false, false);
                }
            }
            // store: out[b][q][h*32 + d], f32
            size_t bwin = (size_t)blockIdx.x * WPB + w;
            for (int ntd = 0; ntd < 2; ++ntd) {
                for (int r = 0; r < 8; ++r) {
                    int q = mt * 16 + r + hi * 8;
                    if (q < NTOK)
                        out[(bwin * NTOK + q) * DIM + h * HD + ntd * 16 + ncol] =
                            ct[ntd][r];
                }
            }
        }
    }
}

extern "C" void kernel_launch(void* const* d_in, const int* in_sizes, int n_in,
                              void* d_out, int out_size, void* d_ws, size_t ws_size,
                              hipStream_t stream) {
    const float* X     = (const float*)d_in[0];
    const float* Wq    = (const float*)d_in[1];
    const float* bq    = (const float*)d_in[2];
    const float* Wk    = (const float*)d_in[3];
    const float* bk    = (const float*)d_in[4];
    const float* Wv    = (const float*)d_in[5];
    const float* bv    = (const float*)d_in[6];
    const float* table = (const float*)d_in[7];
    const int*   idx   = (const int*)d_in[8];
    float* out = (float*)d_out;

    _Float16* Wp   = (_Float16*)d_ws;                        // 1.5 MB
    float*    bexp = (float*)((char*)d_ws + 3 * 32 * 16 * 512 * 2); // 256 KB

    pack_weights<<<(3 * 32 * 16 * 512 + 255) / 256, 256, 0, stream>>>(Wq, Wk, Wv, Wp);
    expand_bias<<<(HEADS * NPAD * NPAD + 255) / 256, 256, 0, stream>>>(table, idx, bexp);

    size_t lds = (size_t)WPB * 4 * 16 * 512 * 2 + 8 * 7168 * 2;  // 240 KB
    swin_attn<<<NWIN / WPB, 256, lds, stream>>>(X, bq, bk, bv, Wp, bexp, out);
}
